// TensorFusion_18073222381568
// MI455X (gfx1250) — compile-verified, matrix operand order checked by reading, not run on hardware
//
#include <hip/hip_runtime.h>
#include <hip/hip_bf16.h>

// ---------------------------------------------------------------------------
// out[n,o] = relu( sum_{a,b,c} x0[n,a]*x1[n,b]*x2[n,c]*W[a,b,c,o] + bias[o] )
// N=2048, D0=D1=33, D2=65, O=128.
// GEMM out = F * W, F built on the fly. W pre-packed once into f16 WMMA
// B-fragment layout in d_ws. Split-K x8 into f32 partials for occupancy
// (512 WGs of 4 waves); wave = 16 rows x 64 cols (4 accumulators) so one
// VALU A-fragment build feeds 8 WMMAs; A build is software-pipelined one
// pair ahead to kill the WMMA->VALU WAR hazard NOPs.
// ---------------------------------------------------------------------------

typedef __attribute__((ext_vector_type(16))) _Float16 v16h;
typedef __attribute__((ext_vector_type(8)))  _Float16 v8h;
typedef __attribute__((ext_vector_type(8)))  float    v8f;

#define NROWS 2048
#define D0 33
#define D1 33
#define D2 65
#define OUTC 128
#define NPAIR (D0 * D1)              // 1089 (a,b) pairs
#define KMAIN (NPAIR * 64)           // 69696: c = 0..63 of every pair
#define STEPS_MAIN (NPAIR * 2)       // 2178 K-steps of 32
#define STEPS_TAIL 35                // 1089 tail elems (c=64) padded to 1120
#define KTAILPAD (STEPS_TAIL * 32)   // 1120
#define STEPS (STEPS_MAIN + STEPS_TAIL) // 2213
#define NJ (OUTC / 16)               // 8 column tiles
#define MROWS 32                     // rows per workgroup (2 row-sets)
#define BLOCK 128                    // 4 waves per workgroup
#define SPLITS 8                     // split-K factor
#define PMAXS 137                    // max pairs per split (last split)

#define WSB_BYTES ((size_t)NJ * STEPS * 32 * 32)            // 18,128,896
#define WSP_BYTES ((size_t)SPLITS * NROWS * OUTC * 4)       //  8,388,608

// ---------------------------------------------------------------------------
// Kernel 1: pack W (f32, [a][b][c][o]) into f16 B fragments, layout [t][j][lane]
// so adjacent column tiles at one K-step are adjacent 1KB blocks.
// Lane holds column o = j*16 + lane%16; K pattern per CDNA5 16-bit fragments:
//   half = lane>=16 ; e<8 : klocal = 8*half + e ; e>=8 : klocal = 16+8*half+e-8
// Global K: k < KMAIN -> pair p = k>>6, c = k&63 ; else tail c=64 slice (padded).
// ---------------------------------------------------------------------------
__global__ void tf_pack_w(const float* __restrict__ W, _Float16* __restrict__ wsB) {
  int g = blockIdx.x * 256 + threadIdx.x;     // one thread per (t, j, lane)
  if (g >= NJ * STEPS * 32) return;
  int lane = g & 31;
  int j    = (g >> 5) & (NJ - 1);
  int t    = g / (NJ * 32);
  int col  = j * 16 + (lane & 15);
  int half = lane >> 4;

  v16h frag;
#pragma unroll
  for (int e = 0; e < 16; ++e) {
    int klocal = (e < 8) ? (8 * half + e) : (16 + 8 * half + (e - 8));
    int k = t * 32 + klocal;
    float val = 0.0f;
    if (k < KMAIN) {
      int p = k >> 6;
      int c = k & 63;
      val = W[((size_t)p * D2 + c) * OUTC + col];
    } else {
      int p2 = k - KMAIN;
      if (p2 < NPAIR) val = W[((size_t)p2 * D2 + 64) * OUTC + col];
    }
    frag[e] = (_Float16)val;
  }
  *reinterpret_cast<v16h*>(wsB + (size_t)g * 16) = frag;   // 32B -> 2x b128
}

// 8 WMMAs of one pair (2 K-steps x 4 column tiles) -------------------------
__device__ __forceinline__ void wmma_pair(const v16h* bp, v16h a0, v16h a1,
                                          v8f& acc0, v8f& acc1,
                                          v8f& acc2, v8f& acc3) {
  v16h b00 = bp[0];
  v16h b01 = bp[32];
  v16h b02 = bp[64];
  v16h b03 = bp[96];
  v16h b10 = bp[NJ * 32];
  v16h b11 = bp[NJ * 32 + 32];
  v16h b12 = bp[NJ * 32 + 64];
  v16h b13 = bp[NJ * 32 + 96];
  acc0 = __builtin_amdgcn_wmma_f32_16x16x32_f16(false, a0, false, b00,
                                                (short)0, acc0, false, false);
  acc1 = __builtin_amdgcn_wmma_f32_16x16x32_f16(false, a0, false, b01,
                                                (short)0, acc1, false, false);
  acc2 = __builtin_amdgcn_wmma_f32_16x16x32_f16(false, a0, false, b02,
                                                (short)0, acc2, false, false);
  acc3 = __builtin_amdgcn_wmma_f32_16x16x32_f16(false, a0, false, b03,
                                                (short)0, acc3, false, false);
  acc0 = __builtin_amdgcn_wmma_f32_16x16x32_f16(false, a1, false, b10,
                                                (short)0, acc0, false, false);
  acc1 = __builtin_amdgcn_wmma_f32_16x16x32_f16(false, a1, false, b11,
                                                (short)0, acc1, false, false);
  acc2 = __builtin_amdgcn_wmma_f32_16x16x32_f16(false, a1, false, b12,
                                                (short)0, acc2, false, false);
  acc3 = __builtin_amdgcn_wmma_f32_16x16x32_f16(false, a1, false, b13,
                                                (short)0, acc3, false, false);
}

// ---------------------------------------------------------------------------
// Kernel 2: split-K fused GEMM. Block = (tile, split); 4 waves cover 32 rows
// x 128 cols (wave = row-set rs of 16 rows, col-group cg of 64 cols with 4
// accumulators). Split covers a contiguous pair range; last split adds tail.
// ---------------------------------------------------------------------------
__global__ void __launch_bounds__(BLOCK)
tf_gemm(const float* __restrict__ x0, const float* __restrict__ x1,
        const float* __restrict__ x2, const _Float16* __restrict__ wsB,
        float* __restrict__ partial) {
  __shared__ __align__(16) float    x0s[MROWS * D0];
  __shared__ __align__(16) float    x1s[MROWS * D1];
  __shared__ __align__(16) float    t2s[MROWS];              // x2[n, 64]
  __shared__ __align__(16) _Float16 x2h[MROWS * 64];         // x2[n,0..63] f16
  __shared__ __align__(16) _Float16 sW[MROWS * PMAXS];       // s_ab per row
  __shared__ __align__(16) _Float16 sprod[MROWS * KTAILPAD]; // tail A, padded

  const int tid   = threadIdx.x;
  const int tile  = blockIdx.x >> 3;           // 0..63
  const int split = blockIdx.x & (SPLITS - 1); // 0..7
  const int n0    = tile * MROWS;
  const int p0    = (split * NPAIR) / SPLITS;
  const int p1    = ((split + 1) * NPAIR) / SPLITS;
  const int np    = p1 - p0;                   // 136.. / 137 last

  // ---- stage x0/x1/x2 tiles into LDS -------------------------------------
  for (int i = tid; i < MROWS * D0; i += BLOCK) {
    int r = i / D0, c = i - r * D0;
    x0s[i] = x0[(size_t)(n0 + r) * D0 + c];
    x1s[i] = x1[(size_t)(n0 + r) * D1 + c];
  }
  for (int i = tid; i < MROWS * D2; i += BLOCK) {
    int r = i / D2, c = i - r * D2;
    float v = x2[(size_t)(n0 + r) * D2 + c];
    if (c < 64) x2h[r * 64 + c] = (_Float16)v;
    else        t2s[r] = v;
  }
  __syncthreads();

  // ---- per-row scale table for this split: sW[r][q] = x0[r,a]*x1[r,b] -----
  for (int i = tid; i < MROWS * PMAXS; i += BLOCK) {
    int r = i / PMAXS, q = i - r * PMAXS;
    if (q < np) {
      int p = p0 + q;
      int a = p / D1, b = p - a * D1;
      sW[i] = (_Float16)(x0s[r * D0 + a] * x1s[r * D1 + b]);
    }
  }
  // ---- tail A (last split only): sprod[r][p] = x0*x1*x2[r,64], padded -----
  if (split == SPLITS - 1) {
    for (int i = tid; i < MROWS * KTAILPAD; i += BLOCK) {
      int r = i / KTAILPAD, p = i - r * KTAILPAD;
      _Float16 v = (_Float16)0.0f;
      if (p < NPAIR) {
        int a = p / D1, b = p - a * D1;
        v = (_Float16)(x0s[r * D0 + a] * x1s[r * D1 + b] * t2s[r]);
      }
      sprod[i] = v;
    }
  }
  __syncthreads();

  const int lane = tid & 31;               // wave32 lane
  const int wv   = tid >> 5;               // wave id 0..3
  const int rs   = wv & 1;                 // row-set (16 rows each)
  const int cg   = wv >> 1;                // column group (64 cols each)
  const int row  = (lane & 15) + rs * 16;  // this lane's source row in tile
  const int half = lane >> 4;

  // ---- hoist x2 A-fragments (loop invariant), documented lane/K pattern ---
  const _Float16* xr = &x2h[row * 64];
  v8h lo0 = *(const v8h*)(xr +      8 * half);
  v8h hi0 = *(const v8h*)(xr + 16 + 8 * half);
  v8h lo1 = *(const v8h*)(xr + 32 + 8 * half);
  v8h hi1 = *(const v8h*)(xr + 48 + 8 * half);
  v16h x2f0 = __builtin_shufflevector(lo0, hi0, 0,1,2,3,4,5,6,7,8,9,10,11,12,13,14,15);
  v16h x2f1 = __builtin_shufflevector(lo1, hi1, 0,1,2,3,4,5,6,7,8,9,10,11,12,13,14,15);

  // packed B fragment stream: layout [t][j][lane]; this wave uses tiles
  // j0 = cg*4 .. j0+3 (adjacent 1KB blocks at each step t).
  const v16h* bp = reinterpret_cast<const v16h*>(wsB)
                 + (size_t)(2 * p0) * (NJ * 32) + (size_t)(cg * 4) * 32 + lane;

  v8f acc0 = {0.f, 0.f, 0.f, 0.f, 0.f, 0.f, 0.f, 0.f};
  v8f acc1 = {0.f, 0.f, 0.f, 0.f, 0.f, 0.f, 0.f, 0.f};
  v8f acc2 = {0.f, 0.f, 0.f, 0.f, 0.f, 0.f, 0.f, 0.f};
  v8f acc3 = {0.f, 0.f, 0.f, 0.f, 0.f, 0.f, 0.f, 0.f};

  // ---- main K loop, A build software-pipelined one pair ahead -------------
  const _Float16* sp = &sW[row * PMAXS];
  _Float16 sh = sp[0];
  v16h a0 = x2f0 * sh;                       // 8x v_pk_mul_f16
  v16h a1 = x2f1 * sh;
#pragma unroll 2
  for (int i = 0; i < np - 1; ++i) {
    _Float16 shn = sp[i + 1];                // next pair's scale
    v16h a0n = x2f0 * shn;                   // fresh regs: no WAR vs WMMAs
    v16h a1n = x2f1 * shn;
    wmma_pair(bp, a0, a1, acc0, acc1, acc2, acc3);
    a0 = a0n;
    a1 = a1n;
    bp += 2 * NJ * 32;                       // advance 2 K-steps
  }
  wmma_pair(bp, a0, a1, acc0, acc1, acc2, acc3);  // final pair (peeled)
  bp += 2 * NJ * 32;

  // ---- tail K loop (last split only; bp already at step 2178) -------------
  if (split == SPLITS - 1) {
    const _Float16* sr0 = &sprod[row * KTAILPAD + 8 * half];
#pragma unroll 5
    for (int tt = 0; tt < STEPS_TAIL; ++tt) {
      const _Float16* sr = sr0 + tt * 32;
      v8h lo = *(const v8h*)(sr);
      v8h hi = *(const v8h*)(sr + 16);
      v16h at = __builtin_shufflevector(lo, hi, 0,1,2,3,4,5,6,7,8,9,10,11,12,13,14,15);
      v16h bt0 = bp[0];
      v16h bt1 = bp[32];
      v16h bt2 = bp[64];
      v16h bt3 = bp[96];
      acc0 = __builtin_amdgcn_wmma_f32_16x16x32_f16(false, at, false, bt0,
                                                    (short)0, acc0, false, false);
      acc1 = __builtin_amdgcn_wmma_f32_16x16x32_f16(false, at, false, bt1,
                                                    (short)0, acc1, false, false);
      acc2 = __builtin_amdgcn_wmma_f32_16x16x32_f16(false, at, false, bt2,
                                                    (short)0, acc2, false, false);
      acc3 = __builtin_amdgcn_wmma_f32_16x16x32_f16(false, at, false, bt3,
                                                    (short)0, acc3, false, false);
      bp += NJ * 32;
    }
  }

  // ---- store f32 partial (C/D layout: m = v + 8*half, col = lane%16) ------
  float* pout = partial + (size_t)split * NROWS * OUTC;
  int colg = cg * 64 + (lane & 15);
  int rowg = n0 + rs * 16 + 8 * half;
#pragma unroll
  for (int v = 0; v < 8; ++v) {
    float* prow = pout + (size_t)(rowg + v) * OUTC + colg;
    prow[0]  = acc0[v];
    prow[16] = acc1[v];
    prow[32] = acc2[v];
    prow[48] = acc3[v];
  }
}

// ---------------------------------------------------------------------------
// Kernel 3: reduce split-K partials, add bias, relu.
// ---------------------------------------------------------------------------
__global__ void tf_reduce(const float* __restrict__ partial,
                          const float* __restrict__ bias,
                          float* __restrict__ out) {
  int i = blockIdx.x * 256 + threadIdx.x;
  if (i >= NROWS * OUTC) return;
  float s = bias[i & (OUTC - 1)];
#pragma unroll
  for (int sp = 0; sp < SPLITS; ++sp)
    s += partial[(size_t)sp * NROWS * OUTC + i];
  out[i] = s > 0.0f ? s : 0.0f;
}

// ---------------------------------------------------------------------------
extern "C" void kernel_launch(void* const* d_in, const int* in_sizes, int n_in,
                              void* d_out, int out_size, void* d_ws, size_t ws_size,
                              hipStream_t stream) {
  const float* x0   = (const float*)d_in[0];
  const float* x1   = (const float*)d_in[1];
  const float* x2   = (const float*)d_in[2];
  const float* W    = (const float*)d_in[3];
  const float* bias = (const float*)d_in[4];
  float* out        = (float*)d_out;
  _Float16* wsB     = (_Float16*)d_ws;
  float* wsP        = (float*)((char*)d_ws + WSB_BYTES);

  if (ws_size < WSB_BYTES + WSP_BYTES) return;  // workspace too small

  // 1) pack W into B-fragment layout (one thread per fragment-lane)
  int packThreads = NJ * STEPS * 32;                 // 566,528
  tf_pack_w<<<(packThreads + 255) / 256, 256, 0, stream>>>(W, wsB);

  // 2) split-K GEMM: 64 tiles x 8 splits = 512 WGs of 4 waves
  tf_gemm<<<(NROWS / MROWS) * SPLITS, BLOCK, 0, stream>>>(x0, x1, x2, wsB, wsP);

  // 3) reduce partials + bias + relu
  tf_reduce<<<(NROWS * OUTC) / 256, 256, 0, stream>>>(wsP, bias, out);
}
